// EfficientTransformer_6176162972205
// MI455X (gfx1250) — matrix operand order repaired and checked by reading, not compiled, not run on hardware
//
#include <hip/hip_runtime.h>
#include <hip/hip_bf16.h>

// ---------------------------------------------------------------------------
// MI455X (gfx1250) axial-attention transformer block.
// All GEMMs: V_WMMA_F32_16X16X32_BF16 (wave32, f32 accumulate), fed by the
// CDNA5 async global->LDS DMA path (GLOBAL_LOAD_ASYNC_TO_LDS_B128 + ASYNCcnt)
// with double-buffered LDS software pipelining.
// ---------------------------------------------------------------------------

typedef __attribute__((ext_vector_type(16))) __bf16 v16bf;
typedef __attribute__((ext_vector_type(8)))  __bf16 bf16x8;
typedef __attribute__((ext_vector_type(8)))  float  v8f;
typedef int i32x4 __attribute__((vector_size(16)));   // matches builtin param type

#define LTOK   110592     // H*W*T
#define CDIM   192
#define NQKV   576
#define HIDD   768
#define NTOK   2304       // H*W   (attention sequence length)
#define TT     48         // T     (folded batch)
#define NHEAD  6
#define DHEAD  1536       // TT * 32  (batch folded into head dim)
#define ATT_SCALE 0.17677669529663689f   // 32^-0.5

#define AS1 __attribute__((address_space(1)))
#define AS3 __attribute__((address_space(3)))

// ---- CDNA5 async global->LDS copy (ASYNCcnt-tracked), builtin or inline asm ----
__device__ __forceinline__ void async_copy_b128(void* lds, const void* gl)
{
#if defined(__has_builtin) && __has_builtin(__builtin_amdgcn_global_load_async_to_lds_b128)
    __builtin_amdgcn_global_load_async_to_lds_b128(
        (AS1 i32x4*)gl, (AS3 i32x4*)lds, 0, 0);
#else
    unsigned l = (unsigned)(unsigned long long)(AS3 char*)lds;
    unsigned long long g = (unsigned long long)gl;
    asm volatile("global_load_async_to_lds_b128 %0, %1, off" :: "v"(l), "v"(g) : "memory");
#endif
}

__device__ __forceinline__ void wait_async_zero()
{
#if defined(__has_builtin) && __has_builtin(__builtin_amdgcn_s_wait_asynccnt)
    __builtin_amdgcn_s_wait_asynccnt(0);
#else
    asm volatile("s_wait_asynccnt 0x0" ::: "memory");
#endif
}

// ===========================================================================
// Generic bf16 WMMA GEMM:  out[M,N] = A[M,K] @ op(B) (+bias)(+gelu)(+resid)
//   BMODE 0: B is (N x K) row-major  (A @ B^T — weight / K^T style)
//   BMODE 1: B is (K x N) row-major  (A @ B    — P @ V style)
//   EPI   0: f32 out, raw                       (attention logits)
//   EPI   1: bf16 out, + optional bias          (QKV, P@V)
//   EPI   2: bf16 out, + bias, exact GELU       (FC1)
//   EPI   3: f32 out, + bias, + f32 residual    (proj -> x1, FC2 -> d_out)
// Block tile 128x64, BK=64, 8 waves; wave owns 32x32 (4 accums, 8 WMMA/stage).
// Double-buffered LDS, async-DMA staged, one barrier per K stage.
// All M % 128 == 0, N % 64 == 0, K % 64 == 0 in this problem.
// ===========================================================================
template<int BMODE, int EPI>
__global__ __launch_bounds__(256)
void gemm_bf16_wmma(const __bf16* __restrict__ A, const __bf16* __restrict__ B,
                    const float* __restrict__ bias, const float* __restrict__ resid,
                    void* __restrict__ out, int M, int N, int K)
{
    constexpr int BM = 128, BN = 64, BK = 64, PITCH = BK + 8;   // 72 elems = 144B rows
    __shared__ __bf16 As[2][BM][PITCH];
    __shared__ __bf16 Bs[2][BN][PITCH];

    const int tid  = threadIdx.x;
    const int lane = tid & 31;
    const int wid  = tid >> 5;
    const int wm   = wid >> 1;          // wave row 0..3 (32 rows each)
    const int wn   = wid & 1;           // wave col 0..1 (32 cols each)
    const int half = lane >> 4;         // K-half select per ISA fragment layout
    const int l16  = lane & 15;
    const int m0   = blockIdx.x * BM;
    const int n0   = blockIdx.y * BN;

    // ---- stage one BK=64 tile (async DMA; BMODE1 B-path transposes via VGPR) --
    auto issue_tile = [&](int kt, int buf) {
        const int k0 = kt * BK;
        // A tile 128x64: 1024 16B vectors over 256 threads
#pragma unroll
        for (int it = 0; it < 4; ++it) {
            int v  = tid + it * 256;
            int r  = v >> 3;
            int kv = (v & 7) * 8;
            async_copy_b128(&As[buf][r][kv],
                            &A[(size_t)(m0 + r) * K + k0 + kv]);
        }
        if (BMODE == 0) {               // B is N x K: rows already k-contiguous
#pragma unroll
            for (int it = 0; it < 2; ++it) {
                int v  = tid + it * 256;
                int r  = v >> 3;
                int kv = (v & 7) * 8;
                async_copy_b128(&Bs[buf][r][kv],
                                &B[(size_t)(n0 + r) * K + k0 + kv]);
            }
        } else {                        // B is K x N: transpose on the way in
#pragma unroll
            for (int it = 0; it < 2; ++it) {
                int v  = tid + it * 256;
                int kk = v >> 3;
                int g  = v & 7;
                bf16x8 vv = *(const bf16x8*)&B[(size_t)(k0 + kk) * N + n0 + g * 8];
#pragma unroll
                for (int e = 0; e < 8; ++e) Bs[buf][g * 8 + e][kk] = vv[e];
            }
        }
    };

    v8f acc[2][2] = {};
    const int ntiles = K / BK;

    issue_tile(0, 0);
    for (int kt = 0; kt < ntiles; ++kt) {
        const int cb = kt & 1;
        wait_async_zero();              // my async stores for tile kt are in LDS
        __syncthreads();                // whole tile visible; prev buffer free
        if (kt + 1 < ntiles) issue_tile(kt + 1, cb ^ 1);   // overlap with compute

        // ---- gather fragments per CDNA5 16-bit A/B VGPR layouts ----
        v16bf af[2][2], bfr[2][2];      // [ks 32-chunk][tile]
#pragma unroll
        for (int ks = 0; ks < 2; ++ks) {
#pragma unroll
            for (int t = 0; t < 2; ++t) {
                int row = wm * 32 + t * 16 + l16;
                bf16x8 lo = *(const bf16x8*)&As[cb][row][ks * 32 + half * 8];
                bf16x8 hi = *(const bf16x8*)&As[cb][row][ks * 32 + 16 + half * 8];
#pragma unroll
                for (int j = 0; j < 8; ++j) { af[ks][t][j] = lo[j]; af[ks][t][8 + j] = hi[j]; }
            }
#pragma unroll
            for (int t = 0; t < 2; ++t) {
                int col = wn * 32 + t * 16 + l16;
                bf16x8 lo = *(const bf16x8*)&Bs[cb][col][ks * 32 + half * 8];
                bf16x8 hi = *(const bf16x8*)&Bs[cb][col][ks * 32 + 16 + half * 8];
#pragma unroll
                for (int j = 0; j < 8; ++j) { bfr[ks][t][j] = lo[j]; bfr[ks][t][8 + j] = hi[j]; }
            }
        }
#pragma unroll
        for (int ks = 0; ks < 2; ++ks)
#pragma unroll
            for (int tm = 0; tm < 2; ++tm)
#pragma unroll
                for (int tn = 0; tn < 2; ++tn)
                    acc[tm][tn] = __builtin_amdgcn_wmma_f32_16x16x32_bf16(
                        false, af[ks][tm], false, bfr[ks][tn],
                        (short)0, acc[tm][tn], false, false);
    }

    // ---- epilogue: D element r sits at (M = r + 8*half, N = l16) ----
#pragma unroll
    for (int tm = 0; tm < 2; ++tm) {
#pragma unroll
        for (int tn = 0; tn < 2; ++tn) {
            int gcol = n0 + wn * 32 + tn * 16 + l16;
            float bv = (EPI == 0) ? 0.0f : (bias ? bias[gcol] : 0.0f);
#pragma unroll
            for (int r = 0; r < 8; ++r) {
                int grow = m0 + wm * 32 + tm * 16 + r + 8 * half;
                size_t o = (size_t)grow * N + gcol;
                float v = acc[tm][tn][r] + bv;
                if (EPI == 2)                         // exact GELU
                    v = 0.5f * v * (1.0f + erff(v * 0.70710678118654752f));
                if (EPI == 3) v += resid[o];
                if (EPI == 0 || EPI == 3) ((float*)out)[o] = v;
                else                      ((__bf16*)out)[o] = (__bf16)v;
            }
        }
    }
}

// ===========================================================================
// LayerNorm (f32 in -> bf16 out), one wave32 per 192-wide row, 8 rows/block.
// permute=1 applies the axial (H,W,T)->(T,H,W) row permutation before QKV.
// ===========================================================================
__global__ __launch_bounds__(256)
void ln_kernel(const float* __restrict__ x, const float* __restrict__ g,
               const float* __restrict__ b, __bf16* __restrict__ out, int permute)
{
    int lane = threadIdx.x & 31;
    int w    = threadIdx.x >> 5;
    int ro   = blockIdx.x * 8 + w;                 // output row
    int ri;
    if (permute) { int t = ro / NTOK, hw = ro % NTOK; ri = hw * TT + t; }
    else         ri = ro;

    const float* xr = x + (size_t)ri * CDIM;
    float vals[6], s = 0.0f;
#pragma unroll
    for (int e = 0; e < 6; ++e) { vals[e] = xr[lane + 32 * e]; s += vals[e]; }
#pragma unroll
    for (int o = 16; o > 0; o >>= 1) s += __shfl_xor(s, o, 32);
    float mean = s * (1.0f / 192.0f);
    float vs = 0.0f;
#pragma unroll
    for (int e = 0; e < 6; ++e) { float d = vals[e] - mean; vs += d * d; }
#pragma unroll
    for (int o = 16; o > 0; o >>= 1) vs += __shfl_xor(vs, o, 32);
    float rstd = rsqrtf(vs * (1.0f / 192.0f) + 1e-5f);
#pragma unroll
    for (int e = 0; e < 6; ++e) {
        int c = lane + 32 * e;
        out[(size_t)ro * CDIM + c] = (__bf16)((vals[e] - mean) * rstd * g[c] + b[c]);
    }
}

// ===========================================================================
// Row softmax over N=2304 logits (f32 in -> bf16 probs out). One block/row.
// ===========================================================================
__global__ __launch_bounds__(256)
void softmax_kernel(const float* __restrict__ S, __bf16* __restrict__ P, int N)
{
    __shared__ float red[256];
    int row = blockIdx.x, tid = threadIdx.x;
    const float* sr = S + (size_t)row * N;
    float v[9];
    float m = -3.0e38f;
#pragma unroll
    for (int i = 0; i < 9; ++i) { v[i] = sr[tid + 256 * i]; m = fmaxf(m, v[i]); }
    red[tid] = m; __syncthreads();
    for (int o = 128; o > 0; o >>= 1) {
        if (tid < o) red[tid] = fmaxf(red[tid], red[tid + o]);
        __syncthreads();
    }
    m = red[0]; __syncthreads();
    float s = 0.0f;
#pragma unroll
    for (int i = 0; i < 9; ++i) { v[i] = __expf(v[i] - m); s += v[i]; }
    red[tid] = s; __syncthreads();
    for (int o = 128; o > 0; o >>= 1) {
        if (tid < o) red[tid] += red[tid + o];
        __syncthreads();
    }
    float inv = 1.0f / red[0];
#pragma unroll
    for (int i = 0; i < 9; ++i)
        P[(size_t)row * N + tid + 256 * i] = (__bf16)(v[i] * inv);
}

// ===========================================================================
// Repack QKV (rows = t*2304+n, cols = 576) into per-head Q/K/V [h][n][d],
// reproducing the module's batch-into-head reshape:
//   flat f = (three*6+h)*1536 + d  ->  (b = f/576, j = f%576)
// Q gets the attention scale folded in.
// ===========================================================================
__global__ __launch_bounds__(256)
void repack_qkv(const __bf16* __restrict__ QKV, __bf16* __restrict__ Q,
                __bf16* __restrict__ Kh, __bf16* __restrict__ V)
{
    size_t idx = (size_t)blockIdx.x * 256 + threadIdx.x;  // 3*6*2304*1536 total
    int d      = (int)(idx % DHEAD);
    size_t t1  = idx / DHEAD;
    int n      = (int)(t1 % NTOK);
    size_t t2  = t1 / NTOK;
    int hh     = (int)(t2 % NHEAD);
    int three  = (int)(t2 / NHEAD);
    int f = (three * NHEAD + hh) * DHEAD + d;
    int b = f / NQKV, j = f % NQKV;
    __bf16 val = QKV[((size_t)b * NTOK + n) * NQKV + j];
    size_t dst = ((size_t)hh * NTOK + n) * DHEAD + d;
    if      (three == 0) Q[dst]  = (__bf16)((float)val * ATT_SCALE);
    else if (three == 1) Kh[dst] = val;
    else                 V[dst]  = val;
}

// Inverse of o.transpose(0,2,1,3).reshape(B_,N,C): dst flat = n*9216 + h*1536 + d
__global__ __launch_bounds__(256)
void unrepack_o(const __bf16* __restrict__ Oh, __bf16* __restrict__ Obf)
{
    size_t i = (size_t)blockIdx.x * 256 + threadIdx.x;    // LTOK*CDIM total
    int rem  = (int)(i % (NHEAD * DHEAD));
    size_t n = i / (NHEAD * DHEAD);
    int h = rem / DHEAD, d = rem % DHEAD;
    Obf[i] = Oh[((size_t)h * NTOK + n) * DHEAD + d];
}

__global__ __launch_bounds__(256)
void f32_to_bf16_kernel(const float* __restrict__ in, __bf16* __restrict__ out, int n)
{
    int i = blockIdx.x * 256 + threadIdx.x;
    if (i < n) out[i] = (__bf16)in[i];
}

// ===========================================================================
// Launch sequence
// ===========================================================================
extern "C" void kernel_launch(void* const* d_in, const int* in_sizes, int n_in,
                              void* d_out, int out_size, void* d_ws, size_t ws_size,
                              hipStream_t stream)
{
    const float* x       = (const float*)d_in[0];
    const float* norm1_g = (const float*)d_in[1];
    const float* norm1_b = (const float*)d_in[2];
    const float* qkv_w   = (const float*)d_in[3];
    const float* qkv_b   = (const float*)d_in[4];
    const float* proj_w  = (const float*)d_in[5];
    const float* proj_b  = (const float*)d_in[6];
    const float* norm2_g = (const float*)d_in[7];
    const float* norm2_b = (const float*)d_in[8];
    const float* fc1_w   = (const float*)d_in[9];
    const float* fc1_b   = (const float*)d_in[10];
    const float* fc2_w   = (const float*)d_in[11];
    const float* fc2_b   = (const float*)d_in[12];
    float* outp          = (float*)d_out;

    // ---- workspace carve-up (256B aligned) ----
    char* base = (char*)d_ws;
    size_t off = 0;
    auto carve = [&](size_t bytes) -> char* {
        off = (off + 255) & ~(size_t)255;
        char* p = base + off;
        off += bytes;
        return p;
    };
    __bf16* Hp    = (__bf16*)carve((size_t)LTOK * CDIM * 2);   // LN1(permuted); reused as LN2 out
    __bf16* Wqkv  = (__bf16*)carve((size_t)NQKV * CDIM * 2);
    __bf16* Wproj = (__bf16*)carve((size_t)CDIM * CDIM * 2);
    __bf16* Wfc1  = (__bf16*)carve((size_t)HIDD * CDIM * 2);
    __bf16* Wfc2  = (__bf16*)carve((size_t)CDIM * HIDD * 2);
    __bf16* QKVb  = (__bf16*)carve((size_t)LTOK * NQKV * 2);   // reused (with Qh) as fc1 out
    __bf16* Qh    = (__bf16*)carve((size_t)NHEAD * NTOK * DHEAD * 2);
    __bf16* Kh    = (__bf16*)carve((size_t)NHEAD * NTOK * DHEAD * 2);
    __bf16* Vh    = (__bf16*)carve((size_t)NHEAD * NTOK * DHEAD * 2);
    float*  Sbuf  = (float*) carve((size_t)NTOK * NTOK * 4);   // one head at a time
    __bf16* Pbuf  = (__bf16*)carve((size_t)NTOK * NTOK * 2);
    __bf16* Ohb   = (__bf16*)carve((size_t)NHEAD * NTOK * DHEAD * 2);
    __bf16* Obf   = (__bf16*)carve((size_t)LTOK * CDIM * 2);
    float*  x1    = (float*) carve((size_t)LTOK * CDIM * 4);
    __bf16* Ln2   = Hp;                                        // Hp dead after QKV GEMM
    __bf16* Fc1o  = QKVb;                                      // QKVb+Qh dead after attention

    // ---- weights f32 -> bf16 ----
    f32_to_bf16_kernel<<<(NQKV * CDIM + 255) / 256, 256, 0, stream>>>(qkv_w, Wqkv, NQKV * CDIM);
    f32_to_bf16_kernel<<<(CDIM * CDIM + 255) / 256, 256, 0, stream>>>(proj_w, Wproj, CDIM * CDIM);
    f32_to_bf16_kernel<<<(HIDD * CDIM + 255) / 256, 256, 0, stream>>>(fc1_w, Wfc1, HIDD * CDIM);
    f32_to_bf16_kernel<<<(CDIM * HIDD + 255) / 256, 256, 0, stream>>>(fc2_w, Wfc2, CDIM * HIDD);

    // ---- LN1 + axial permute ----
    ln_kernel<<<LTOK / 8, 256, 0, stream>>>(x, norm1_g, norm1_b, Hp, 1);

    // ---- QKV projection: (L,192) @ (576,192)^T ----
    gemm_bf16_wmma<0, 1><<<dim3(LTOK / 128, NQKV / 64), 256, 0, stream>>>(
        Hp, Wqkv, qkv_b, nullptr, QKVb, LTOK, NQKV, CDIM);

    // ---- batch-into-head repack (Q pre-scaled) ----
    repack_qkv<<<(3 * NHEAD * NTOK * DHEAD) / 256, 256, 0, stream>>>(QKVb, Qh, Kh, Vh);

    // ---- attention, head-serial: S = Q K^T ; P = softmax(S) ; O = P V ----
    for (int h = 0; h < NHEAD; ++h) {
        const size_t hs = (size_t)h * NTOK * DHEAD;
        gemm_bf16_wmma<0, 0><<<dim3(NTOK / 128, NTOK / 64), 256, 0, stream>>>(
            Qh + hs, Kh + hs, nullptr, nullptr, Sbuf, NTOK, NTOK, DHEAD);
        softmax_kernel<<<NTOK, 256, 0, stream>>>(Sbuf, Pbuf, NTOK);
        gemm_bf16_wmma<1, 1><<<dim3(NTOK / 128, DHEAD / 64), 256, 0, stream>>>(
            Pbuf, Vh + hs, nullptr, nullptr, Ohb + hs, NTOK, DHEAD, NTOK);
    }

    // ---- undo head reshape ----
    unrepack_o<<<(LTOK * CDIM) / 256, 256, 0, stream>>>(Ohb, Obf);

    // ---- output projection + residual (flat, module does NOT undo permute) ----
    gemm_bf16_wmma<0, 3><<<dim3(LTOK / 128, CDIM / 64), 256, 0, stream>>>(
        Obf, Wproj, proj_b, x, x1, LTOK, CDIM, CDIM);

    // ---- MLP: LN2 -> FC1+GELU -> FC2 + residual -> d_out ----
    ln_kernel<<<LTOK / 8, 256, 0, stream>>>(x1, norm2_g, norm2_b, Ln2, 0);
    gemm_bf16_wmma<0, 2><<<dim3(LTOK / 128, HIDD / 64), 256, 0, stream>>>(
        Ln2, Wfc1, fc1_b, nullptr, Fc1o, LTOK, HIDD, CDIM);
    gemm_bf16_wmma<0, 3><<<dim3(LTOK / 128, CDIM / 64), 256, 0, stream>>>(
        Fc1o, Wfc2, fc2_b, x1, outp, LTOK, CDIM, HIDD);
}
